// AttentionModule_83313775608133
// MI455X (gfx1250) — matrix operand order repaired and checked by reading, not compiled, not run on hardware
//
#include <hip/hip_runtime.h>
#include <hip/hip_bf16.h>
#include <math.h>

typedef __attribute__((ext_vector_type(16))) __bf16 v16bf;
typedef __attribute__((ext_vector_type(8)))  float  v8f;

#define BB 128
#define SS 2048
#define HH 512
#define AA 512
#define XK 514   // padded LDS row stride (bf16 elems): 257 dwords -> conflict-free
constexpr float INV_SQRT_A = 0.044194173824159216f; // 1/sqrt(512)

// ---------------- K1: W f32 -> bf16 ----------------
__global__ __launch_bounds__(256) void k_conv_w(const float* __restrict__ w,
                                                __bf16* __restrict__ wbf) {
  int i = blockIdx.x * 256 + threadIdx.x;
  wbf[i] = (__bf16)w[i];
}

// ---------------- K2: fused tanh(X@W+b)/sqrt(A) . u -> vu[B*S] ----------------
// Block: 128 threads (4 waves). Each block: 128 rows of flattened [B*S, H].
// Each wave: 32 rows (2 M-tiles, B-fragment reused across both -> 1 b128/WMMA).
// N looped in 8 groups of 64 (4 tiles of 16), K in steps of 32.
__global__ __launch_bounds__(128)
void k_gemm_tanh_vu(const float* __restrict__ x, const __bf16* __restrict__ wbf,
                    const float* __restrict__ bvec, const float* __restrict__ uvec,
                    float* __restrict__ vu) {
  extern __shared__ __bf16 xs[];   // 128 rows x XK bf16 = 131584 B
  const int tid = threadIdx.x;
  const size_t rowBase = (size_t)blockIdx.x * 128;

  // Stage X tile [128 x 512] f32 -> bf16 into LDS
  const float4* xv = (const float4*)(x + rowBase * HH);
  #pragma unroll 4
  for (int i = tid; i < 128 * HH / 4; i += 128) {
    float4 f = xv[i];
    int row = i >> 7;          // 128 float4 per row
    int k   = (i & 127) << 2;
    union { __bf16 h[2]; unsigned u; } p0, p1;
    p0.h[0] = (__bf16)f.x; p0.h[1] = (__bf16)f.y;
    p1.h[0] = (__bf16)f.z; p1.h[1] = (__bf16)f.w;
    *(unsigned*)(xs + row * XK + k)     = p0.u;
    *(unsigned*)(xs + row * XK + k + 2) = p1.u;
  }
  __syncthreads();

  const int lane = tid & 31;
  const int wave = tid >> 5;
  const int l15  = lane & 15;
  const int g16  = lane >> 4;
  // A-frag rows: lane holds M = lane&15 of each 16-row tile
  const __bf16* xrow0 = xs + (wave * 32 + l15) * XK;
  const __bf16* xrow1 = xs + (wave * 32 + 16 + l15) * XK;

  float vuacc[2][8] = {{0.f,0.f,0.f,0.f,0.f,0.f,0.f,0.f},
                       {0.f,0.f,0.f,0.f,0.f,0.f,0.f,0.f}};

  for (int g = 0; g < 8; ++g) {          // N group: 64 columns
    const int n0 = g * 64;
    v8f acc[2][4];
    #pragma unroll
    for (int j = 0; j < 4; ++j) {
      float bias = bvec[n0 + j * 16 + l15];   // C/D: lane = N
      #pragma unroll
      for (int v = 0; v < 8; ++v) { acc[0][j][v] = bias; acc[1][j][v] = bias; }
    }
    for (int kt = 0; kt < HH; kt += 32) {
      // A fragments (16-bit A 16x32 layout): VGPR v holds K pair
      // kb = g16*8 + (v&3)*2 + (v>>2)*16
      union { v16bf v; unsigned u[8]; } af0, af1;
      #pragma unroll
      for (int vv = 0; vv < 8; ++vv) {
        int kb = kt + g16 * 8 + (vv & 3) * 2 + ((vv >> 2) * 16);
        af0.u[vv] = *(const unsigned*)(xrow0 + kb);
        af1.u[vv] = *(const unsigned*)(xrow1 + kb);
      }
      // B fragment: lane = N column (l15), element s = K row 8*g16 + (s&7) + 16*(s>>3) of W
      const __bf16* wrow = wbf + (size_t)kt * AA + n0 + l15;
      #pragma unroll
      for (int j = 0; j < 4; ++j) {
        v16bf bf;
        #pragma unroll
        for (int s = 0; s < 16; ++s)
          bf[s] = wrow[(size_t)(g16 * 8 + (s & 7) + (s >> 3) * 16) * AA + j * 16];
        acc[0][j] = __builtin_amdgcn_wmma_f32_16x16x32_bf16(
            false, af0.v, false, bf, (short)0, acc[0][j], false, false);
        acc[1][j] = __builtin_amdgcn_wmma_f32_16x16x32_bf16(
            false, af1.v, false, bf, (short)0, acc[1][j], false, false);
      }
    }
    // Epilogue: tanh, scale, dot with u (lane's column), accumulate per row
    #pragma unroll
    for (int j = 0; j < 4; ++j) {
      float un = uvec[n0 + j * 16 + l15];
      #pragma unroll
      for (int v = 0; v < 8; ++v) {
        vuacc[0][v] = fmaf(tanhf(acc[0][j][v]) * INV_SQRT_A, un, vuacc[0][v]);
        vuacc[1][v] = fmaf(tanhf(acc[1][j][v]) * INV_SQRT_A, un, vuacc[1][v]);
      }
    }
  }
  // Butterfly reduce over the 16 lanes holding different N of the same row
  #pragma unroll
  for (int mt = 0; mt < 2; ++mt)
    #pragma unroll
    for (int v = 0; v < 8; ++v) {
      #pragma unroll
      for (int off = 8; off >= 1; off >>= 1)
        vuacc[mt][v] += __shfl_xor(vuacc[mt][v], off, 32);
    }
  if (l15 == 0) {   // lane 0 -> rows M=0..7 of each tile; lane 16 -> M=8..15
    size_t r0 = rowBase + (size_t)wave * 32 + g16 * 8;
    #pragma unroll
    for (int v = 0; v < 8; ++v) {
      vu[r0 + v]      = vuacc[0][v];
      vu[r0 + 16 + v] = vuacc[1][v];
    }
  }
}

// ---------------- K3: softmax over S per batch -> alphas ----------------
__global__ __launch_bounds__(256)
void k_softmax(const float* __restrict__ vu, float* __restrict__ alphas) {
  __shared__ float red[256];
  const int b = blockIdx.x, tid = threadIdx.x;
  const float* vr = vu + (size_t)b * SS;
  float vals[8]; float m = -INFINITY;
  #pragma unroll
  for (int i = 0; i < 8; ++i) { vals[i] = vr[i * 256 + tid]; m = fmaxf(m, vals[i]); }
  red[tid] = m; __syncthreads();
  for (int s = 128; s > 0; s >>= 1) { if (tid < s) red[tid] = fmaxf(red[tid], red[tid + s]); __syncthreads(); }
  m = red[0]; __syncthreads();
  float sum = 0.f;
  #pragma unroll
  for (int i = 0; i < 8; ++i) { vals[i] = expf(vals[i] - m); sum += vals[i]; }
  red[tid] = sum; __syncthreads();
  for (int s = 128; s > 0; s >>= 1) { if (tid < s) red[tid] += red[tid + s]; __syncthreads(); }
  float inv = 1.0f / (red[0] + 1e-10f);
  float* ar = alphas + (size_t)b * SS;
  #pragma unroll
  for (int i = 0; i < 8; ++i) ar[i * 256 + tid] = vals[i] * inv;
}

// ---------------- K4: partial weighted sums  part[sc][b][h] = sum_{s in chunk} a*X ----
__global__ __launch_bounds__(256)
void k_weighted_sum(const float* __restrict__ x, const float* __restrict__ alphas,
                    float* __restrict__ part) {
  __shared__ float sal[512];
  const int tid = threadIdx.x;
  const int hc = blockIdx.x, sc = blockIdx.y, b = blockIdx.z;
  const int s0 = sc * 512;
  sal[tid]       = alphas[(size_t)b * SS + s0 + tid];
  sal[tid + 256] = alphas[(size_t)b * SS + s0 + tid + 256];
  __syncthreads();
  const int h = hc * 256 + tid;
  const float* xp = x + ((size_t)b * SS + s0) * HH + h;
  float acc = 0.f;
  #pragma unroll 8
  for (int s = 0; s < 512; ++s) acc = fmaf(sal[s], xp[(size_t)s * HH], acc);
  part[(size_t)sc * (BB * HH) + (size_t)b * HH + h] = acc;
}

// ---------------- K5: reduce 4 partials -> d_out[0:B*H] ----------------
__global__ __launch_bounds__(256)
void k_reduce_out(const float* __restrict__ part, float* __restrict__ out) {
  int i = blockIdx.x * 256 + threadIdx.x;
  out[i] = part[i] + part[65536 + i] + part[131072 + i] + part[196608 + i];
}

extern "C" void kernel_launch(void* const* d_in, const int* in_sizes, int n_in,
                              void* d_out, int out_size, void* d_ws, size_t ws_size,
                              hipStream_t stream) {
  const float* x    = (const float*)d_in[0];   // [B,S,H]
  const float* w    = (const float*)d_in[1];   // [H,A]
  const float* bvec = (const float*)d_in[2];   // [A]
  const float* uvec = (const float*)d_in[3];   // [A]
  float* out    = (float*)d_out;               // [B*H]
  float* alphas = out + BB * HH;               // [B*S]

  char*   wsb  = (char*)d_ws;
  __bf16* wbf  = (__bf16*)wsb;                           // 512 KB
  float*  vu   = (float*)(wsb + 524288);                 // 1 MB
  float*  part = (float*)(wsb + 524288 + 1048576);       // 1 MB

  k_conv_w<<<(HH * AA) / 256, 256, 0, stream>>>(w, wbf);
  k_gemm_tanh_vu<<<(BB * SS) / 128, 128, 128 * XK * sizeof(__bf16), stream>>>(
      x, wbf, bvec, uvec, vu);
  k_softmax<<<BB, 256, 0, stream>>>(vu, alphas);
  k_weighted_sum<<<dim3(2, 4, BB), 256, 0, stream>>>(x, alphas, part);
  k_reduce_out<<<(BB * HH) / 256, 256, 0, stream>>>(part, out);
}
